// Attention_11785390260995
// MI455X (gfx1250) — compile-verified, matrix operand order checked by reading, not run on hardware
//
#include <hip/hip_runtime.h>
#include <hip/hip_bf16.h>

// ---------------------------------------------------------------------------
// Fused attention: QKV proj -> flash attention (causal + key padding) -> out proj
// MI455X / gfx1250: wave32, v_wmma_f32_16x16x32_bf16, async global->LDS staging.
// ---------------------------------------------------------------------------

#define BATCH   2
#define SEQ     2048
#define DMODEL  1024
#define HEADS   16
#define DHEAD   64
#define INNER   (HEADS * DHEAD)        // 1024
#define NCOL3   (3 * INNER)            // 3072
#define MROWS   (BATCH * SEQ)          // 4096
#define SCALE_F 0.125f                 // 64^-0.5

typedef __attribute__((ext_vector_type(16))) __bf16 v16bf;
typedef __attribute__((ext_vector_type(8)))  float  v8f;

union Frag16 { v16bf v; unsigned u[8]; };

// K-offset of dword i within a 16-bit A/B fragment (CDNA5 ISA 7.12.2):
// lanes 0-15: V0..V3 -> K 0..7, V4..V7 -> K 16..23 ; lanes 16-31: +8.
__device__ __forceinline__ int kOff(int i, int h) {
    return ((i & 4) << 2) + h * 8 + ((i & 3) << 1);
}

// Load one 16x32 bf16 fragment: p = per-lane row/column base pointer (contiguous
// in the contraction dimension), kk = k-step base, h = lane>>4.
__device__ __forceinline__ v16bf load_frag(const __bf16* p, int kk, int h) {
    Frag16 f;
#pragma unroll
    for (int i = 0; i < 8; ++i)
        f.u[i] = *(const unsigned*)(p + kk + kOff(i, h));
    return f.v;
}

__device__ __forceinline__ v8f wmma_bf16(v16bf a, v16bf b, v8f c) {
    return __builtin_amdgcn_wmma_f32_16x16x32_bf16(
        /*neg_a=*/false, a, /*neg_b=*/false, b,
        /*c_mod=*/(short)0, c, /*reuse_a=*/false, /*reuse_b=*/false);
}

// Async copy 16 bytes global -> LDS (gfx1250, tracked by ASYNCcnt).
// Flat addresses into LDS truncate to the LDS offset (ISA 10.2 aperture rules),
// so the low 32 bits of a __shared__ pointer are the DS destination address.
__device__ __forceinline__ void async_copy_b128(void* lds_dst, const void* gsrc) {
    unsigned ldso = (unsigned)(size_t)lds_dst;
    asm volatile("global_load_async_to_lds_b128 %0, %1, off"
                 :: "v"(ldso), "v"(gsrc) : "memory");
}

__device__ __forceinline__ void wait_async0() {
    asm volatile("s_wait_asynccnt 0x0" ::: "memory");
}

// ---------------------------------------------------------------------------
// Conversion kernels
// ---------------------------------------------------------------------------
__global__ void cvt_f32_bf16(const float* __restrict__ src,
                             __bf16* __restrict__ dst, int n) {
    int i = blockIdx.x * blockDim.x + threadIdx.x;
    if (i < n) dst[i] = (__bf16)src[i];
}

// src is rows x cols row-major; dst is cols x rows (column-major view of src).
__global__ void cvt_transpose_bf16(const float* __restrict__ src,
                                   __bf16* __restrict__ dst,
                                   int rows, int cols) {
    int i = blockIdx.x * blockDim.x + threadIdx.x;
    if (i < rows * cols) {
        int r = i / cols, c = i % cols;
        dst[c * rows + r] = (__bf16)src[i];
    }
}

// ---------------------------------------------------------------------------
// QKV GEMM: C[4096 x 3072] = xh[4096 x 1024] @ Wqkv[1024 x 3072]
// Scatter into Q (scaled, (B,H,N,d)), K ((B,H,N,d)), V^T ((B,H,d,N)).
// ---------------------------------------------------------------------------
__global__ __launch_bounds__(256) void qkv_gemm(
        const __bf16* __restrict__ xh, const __bf16* __restrict__ wqkv_t,
        __bf16* __restrict__ Qb, __bf16* __restrict__ Kb,
        __bf16* __restrict__ Vt) {
    const int lane = threadIdx.x & 31, wave = threadIdx.x >> 5;
    const int h = lane >> 4, ln = lane & 15;
    const int m0 = blockIdx.y * 32 + (wave >> 2) * 16;
    const int n0 = blockIdx.x * 256 + (wave & 3) * 64;

    const __bf16* arow = xh + (size_t)(m0 + ln) * DMODEL;
    v8f acc[4] = {};
    for (int kk = 0; kk < DMODEL; kk += 32) {
        if (kk + 64 < DMODEL) __builtin_prefetch(arow + kk + 64, 0, 0);
        v16bf a = load_frag(arow, kk, h);
#pragma unroll
        for (int t = 0; t < 4; ++t) {
            const __bf16* bcol = wqkv_t + (size_t)(n0 + t * 16 + ln) * DMODEL;
            v16bf b = load_frag(bcol, kk, h);
            acc[t] = wmma_bf16(a, b, acc[t]);
        }
    }

    const int which = n0 / INNER;           // 0=Q, 1=K, 2=V (uniform per wave)
    const int hh    = (n0 % INNER) / DHEAD; // head (uniform per wave)
#pragma unroll
    for (int t = 0; t < 4; ++t) {
#pragma unroll
        for (int r = 0; r < 8; ++r) {
            int m = m0 + r + 8 * h;
            int b = m >> 11, nseq = m & (SEQ - 1);
            int dd = t * 16 + ln;
            float v = acc[t][r];
            size_t bh = (size_t)(b * HEADS + hh);
            if (which == 0)
                Qb[(bh * SEQ + nseq) * DHEAD + dd] = (__bf16)(v * SCALE_F);
            else if (which == 1)
                Kb[(bh * SEQ + nseq) * DHEAD + dd] = (__bf16)v;
            else
                Vt[(bh * DHEAD + dd) * SEQ + nseq] = (__bf16)v;
        }
    }
}

// ---------------------------------------------------------------------------
// Flash attention. One block = 8 waves = 128 consecutive query rows of one
// (b,h). Each 32-key step: the whole block async-stages K (32x64) and V^T
// (64x32) tiles into LDS once (global_load_async_to_lds_b128 + ASYNCcnt),
// then each wave runs WMMA S = Q K^T, online softmax, and O += P V from LDS.
// ---------------------------------------------------------------------------
__global__ __launch_bounds__(256) void flash_attn(
        const __bf16* __restrict__ Qb, const __bf16* __restrict__ Kb,
        const __bf16* __restrict__ Vt, const unsigned char* __restrict__ mask,
        __bf16* __restrict__ A2) {
    __shared__ __bf16 Ktile[32 * DHEAD];     // 4 KB: rows j (32) x d (64)
    __shared__ __bf16 Vtile[DHEAD * 32];     // 4 KB: rows d (64) x j (32)
    __shared__ __bf16 pst[8][16 * 32];       // 8 KB: per-wave P staging

    const int tid  = threadIdx.x;
    const int lane = tid & 31, wave = tid >> 5;
    const int h = lane >> 4, ln = lane & 15;
    const int bh  = blockIdx.x / (SEQ / 128);       // (b*H + head)
    const int qb0 = (blockIdx.x % (SEQ / 128)) * 128;
    const int q0  = qb0 + wave * 16;
    const int b   = bh >> 4, hh = bh & 15;

    const __bf16* qrow = Qb + ((size_t)bh * SEQ + q0 + ln) * DHEAD;
    const v16bf qa0 = load_frag(qrow, 0, h);
    const v16bf qa1 = load_frag(qrow, 32, h);
    const unsigned char* mk = mask + b * SEQ;
    __bf16* lds = &pst[wave][0];

    float mrow[8], lrow[8];
    v8f o[4] = {};
#pragma unroll
    for (int r = 0; r < 8; ++r) { mrow[r] = -1e30f; lrow[r] = 0.f; }

    const int jend = qb0 + 112;              // max q0 across the block (causal)
    for (int j0 = 0; j0 <= jend; j0 += 32) {
        __syncthreads();                     // tiles free to overwrite
        // ---- cooperative async staging: 256 thr x 2 b128 = 8 KB ----
        {
            int kr = tid >> 3, kc = tid & 7; // K: 32 rows x 8 chunks of 8 bf16
            async_copy_b128(&Ktile[kr * DHEAD + kc * 8],
                            Kb + ((size_t)bh * SEQ + j0 + kr) * DHEAD + kc * 8);
            int vr = tid >> 2, vc = tid & 3; // V^T: 64 rows x 4 chunks of 8 bf16
            async_copy_b128(&Vtile[vr * 32 + vc * 8],
                            Vt + ((size_t)bh * DHEAD + vr) * SEQ + j0 + vc * 8);
        }
        wait_async0();
        __syncthreads();                     // tiles visible to all waves

        if (j0 <= q0) {                      // causal: this wave needs the tile
            // ---- S = Q K^T for 32 keys (two 16x16 tiles) ----
            v8f s[2] = {};
#pragma unroll
            for (int t = 0; t < 2; ++t) {
                const __bf16* krow = &Ktile[(t * 16 + ln) * DHEAD];
                v16bf kb0 = load_frag(krow, 0, h);
                v16bf kb1 = load_frag(krow, 32, h);
                s[t] = wmma_bf16(qa0, kb0, s[t]);
                s[t] = wmma_bf16(qa1, kb1, s[t]);
            }
            // ---- causal + key-padding mask ----
            int mv[2];
#pragma unroll
            for (int t = 0; t < 2; ++t) mv[t] = mk[j0 + t * 16 + ln];
#pragma unroll
            for (int t = 0; t < 2; ++t) {
                int jcol = j0 + t * 16 + ln;
#pragma unroll
                for (int r = 0; r < 8; ++r) {
                    int qr = q0 + r + 8 * h;
                    if (jcol > qr || !mv[t]) s[t][r] = -1e30f;
                }
            }
            // ---- online softmax row stats (16-lane reductions) ----
            float alpha[8];
#pragma unroll
            for (int r = 0; r < 8; ++r) {
                float x = fmaxf(s[0][r], s[1][r]);
#pragma unroll
                for (int off = 8; off; off >>= 1)
                    x = fmaxf(x, __shfl_xor(x, off, 16));
                float mn = fmaxf(mrow[r], x);
                alpha[r] = __expf(mrow[r] - mn);
                mrow[r] = mn;
                float p0 = __expf(s[0][r] - mn);
                float p1 = __expf(s[1][r] - mn);
                s[0][r] = p0; s[1][r] = p1;
                float ss = p0 + p1;
#pragma unroll
                for (int off = 8; off; off >>= 1)
                    ss += __shfl_xor(ss, off, 16);
                lrow[r] = lrow[r] * alpha[r] + ss;
#pragma unroll
                for (int t = 0; t < 4; ++t) o[t][r] *= alpha[r];
            }
            // ---- stage P (C layout -> row-major LDS -> A layout) ----
#pragma unroll
            for (int t = 0; t < 2; ++t)
#pragma unroll
                for (int r = 0; r < 8; ++r)
                    lds[(r + 8 * h) * 32 + t * 16 + ln] = (__bf16)s[t][r];
            asm volatile("s_wait_dscnt 0x0" ::: "memory");
            v16bf pa = load_frag(lds + ln * 32, 0, h);
            // ---- O += P V from LDS V^T tile ----
#pragma unroll
            for (int t = 0; t < 4; ++t) {
                const __bf16* vrp = &Vtile[(t * 16 + ln) * 32];
                v16bf vb = load_frag(vrp, 0, h);
                o[t] = wmma_bf16(pa, vb, o[t]);
            }
        }
    }

    // ---- normalize and store into (B*N, H*d) bf16 ----
#pragma unroll
    for (int r = 0; r < 8; ++r) {
        float inv = lrow[r] > 0.f ? 1.f / lrow[r] : 0.f;
        int gm = b * SEQ + q0 + r + 8 * h;
#pragma unroll
        for (int t = 0; t < 4; ++t)
            A2[(size_t)gm * INNER + hh * DHEAD + t * 16 + ln] =
                (__bf16)(o[t][r] * inv);
    }
}

// ---------------------------------------------------------------------------
// Output GEMM: out[4096 x 1024] = A2[4096 x 1024] @ Wout[1024 x 1024] + b_out
// ---------------------------------------------------------------------------
__global__ __launch_bounds__(256) void out_gemm(
        const __bf16* __restrict__ A2, const __bf16* __restrict__ wout_t,
        const float* __restrict__ bout, float* __restrict__ out) {
    const int lane = threadIdx.x & 31, wave = threadIdx.x >> 5;
    const int h = lane >> 4, ln = lane & 15;
    const int m0 = blockIdx.y * 32 + (wave >> 2) * 16;
    const int n0 = blockIdx.x * 256 + (wave & 3) * 64;

    const __bf16* arow = A2 + (size_t)(m0 + ln) * INNER;
    v8f acc[4] = {};
    for (int kk = 0; kk < INNER; kk += 32) {
        if (kk + 64 < INNER) __builtin_prefetch(arow + kk + 64, 0, 0);
        v16bf a = load_frag(arow, kk, h);
#pragma unroll
        for (int t = 0; t < 4; ++t) {
            const __bf16* bcol = wout_t + (size_t)(n0 + t * 16 + ln) * INNER;
            v16bf b = load_frag(bcol, kk, h);
            acc[t] = wmma_bf16(a, b, acc[t]);
        }
    }
#pragma unroll
    for (int t = 0; t < 4; ++t) {
        int col = n0 + t * 16 + ln;
        float bias = bout[col];
#pragma unroll
        for (int r = 0; r < 8; ++r) {
            int m = m0 + r + 8 * h;
            out[(size_t)m * DMODEL + col] = acc[t][r] + bias;
        }
    }
}

// ---------------------------------------------------------------------------
// Host launcher
// ---------------------------------------------------------------------------
extern "C" void kernel_launch(void* const* d_in, const int* in_sizes, int n_in,
                              void* d_out, int out_size, void* d_ws, size_t ws_size,
                              hipStream_t stream) {
    const float*         x     = (const float*)d_in[0];
    const unsigned char* mask  = (const unsigned char*)d_in[1];  // jnp bool
    const float*         W_qkv = (const float*)d_in[2];
    const float*         W_out = (const float*)d_in[3];
    const float*         b_out = (const float*)d_in[4];
    float*               out   = (float*)d_out;

    char* ws = (char*)d_ws;
    size_t off = 0;
    __bf16* xh     = (__bf16*)(ws + off); off += (size_t)MROWS * DMODEL * 2;   // 8 MB
    __bf16* wqkv_t = (__bf16*)(ws + off); off += (size_t)NCOL3 * DMODEL * 2;   // 6 MB
    __bf16* wout_t = (__bf16*)(ws + off); off += (size_t)INNER * DMODEL * 2;   // 2 MB
    __bf16* Qb     = (__bf16*)(ws + off); off += (size_t)MROWS * INNER * 2;    // 8 MB
    __bf16* Kb     = (__bf16*)(ws + off); off += (size_t)MROWS * INNER * 2;    // 8 MB
    __bf16* Vt     = (__bf16*)(ws + off); off += (size_t)MROWS * INNER * 2;    // 8 MB
    __bf16* A2     = (__bf16*)(ws + off); off += (size_t)MROWS * INNER * 2;    // 8 MB

    // 1) conversions
    {
        int n = MROWS * DMODEL;
        cvt_f32_bf16<<<(n + 255) / 256, 256, 0, stream>>>(x, xh, n);
        int nq = DMODEL * NCOL3;
        cvt_transpose_bf16<<<(nq + 255) / 256, 256, 0, stream>>>(W_qkv, wqkv_t, DMODEL, NCOL3);
        int no = INNER * DMODEL;
        cvt_transpose_bf16<<<(no + 255) / 256, 256, 0, stream>>>(W_out, wout_t, INNER, DMODEL);
    }
    // 2) QKV projection
    qkv_gemm<<<dim3(NCOL3 / 256, MROWS / 32), 256, 0, stream>>>(xh, wqkv_t, Qb, Kb, Vt);
    // 3) flash attention: 512 blocks x (8 waves = 128 query rows)
    flash_attn<<<BATCH * HEADS * (SEQ / 128), 256, 0, stream>>>(Qb, Kb, Vt, mask, A2);
    // 4) output projection + bias
    out_gemm<<<dim3(DMODEL / 256, MROWS / 32), 256, 0, stream>>>(A2, wout_t, b_out, out);

    (void)in_sizes; (void)n_in; (void)out_size; (void)ws_size;
}
